// VRNN_82824149336186
// MI455X (gfx1250) — matrix operand order repaired and checked by reading, not compiled, not run on hardware
//
#include <hip/hip_runtime.h>

// Problem sizes (compile-time).
#define B_  256
#define T_  128
#define I_  128
#define H_  1024
#define L_  256

// Group structure: 16 groups (16 batch rows each) x 8 N-slice blocks = 128 blocks.
#define NSLICE_   8
#define NGROUPS_  (B_ / 16)
#define NWORKERS_ (NSLICE_ * 8)     // 8 blocks x 8 waves = 64 tile workers / group
#define GTHREADS_ (NSLICE_ * 256)   // 2048 threads / group

typedef __attribute__((ext_vector_type(16))) __bf16 v16bf;
typedef __attribute__((ext_vector_type(8)))  __bf16 v8bf;
typedef __attribute__((ext_vector_type(8)))  float  v8f;

// ---------------------------------------------------------------------------
// Per-group activation buffers in workspace (bf16 region, element offsets).
// All offsets are multiples of 16 elements -> 16B-aligned v8bf loads.
// ---------------------------------------------------------------------------
#define GA_XIN    0
#define GA_HID    (GA_XIN   + 16 * I_)
#define GA_EMBX   (GA_HID   + 16 * H_)
#define GA_EMBZ   (GA_EMBX  + 16 * H_)
#define GA_EMBXH  (GA_EMBZ  + 16 * H_)
#define GA_Z      (GA_EMBXH + 16 * H_)
#define GA_XH     (GA_Z     + 16 * L_)
#define GA_H0     (GA_XH    + 16 * I_)
#define GA_H1     (GA_H0    + 16 * H_)
#define GA_BF16_TOTAL (GA_H1 + 16 * H_)
// f32 region (element offsets)
#define GF_C      0
#define GF_F      (GF_C + 16 * H_)
#define GF_TOTAL  (GF_F + 16 * (2 * L_))
// total bf16 weight elements (N-major transposed)
#define WOFF_TOTAL 20054016

// ---------------------------------------------------------------------------
// Weight conversion: f32 row-major (K x N) -> bf16 N-major (N x K).
// ---------------------------------------------------------------------------
__global__ void wcvt_kernel(const float* __restrict__ src, __bf16* __restrict__ dst,
                            int K, int N) {
    int idx = blockIdx.x * blockDim.x + threadIdx.x;
    if (idx >= K * N) return;
    int n = idx / K;
    int k = idx - n * K;
    dst[idx] = (__bf16)src[(size_t)k * N + n];
}

// ---------------------------------------------------------------------------
// Zero the recurrent state + group barrier counters (every call -> no
// cross-call state, deterministic).
// ---------------------------------------------------------------------------
__global__ void init_ws_kernel(__bf16* actB, float* actF, unsigned* bars) {
    size_t idx = (size_t)blockIdx.x * 256 + threadIdx.x;
    const size_t nstate = (size_t)NGROUPS_ * 16 * H_;
    if (idx < nstate) {
        int g = (int)(idx / (16 * H_));
        size_t o = idx - (size_t)g * (16 * H_);
        actB[(size_t)g * GA_BF16_TOTAL + GA_H0 + o] = (__bf16)0.0f;   // h(t=0)
        actF[(size_t)g * GF_TOTAL + GF_C + o] = 0.0f;                 // c(t=0)
    }
    if (idx < NGROUPS_ * 32) bars[idx] = 0u;
}

// ---------------------------------------------------------------------------
// One WMMA step: acc += A-chunk(16x32) * B-chunk(32x16), bf16->f32.
// a points at (row m, k + kh*8); lanes 0-15 supply K 0-7/16-23, lanes 16-31
// supply K 8-15/24-31 per the ISA 16-bit A layout. b points at (col n, k+kh*8)
// of the N-major weight -> two 16B loads in the same K order as A.
// ---------------------------------------------------------------------------
__device__ __forceinline__ v8f wmma_step(v8f acc, const __bf16* a, const __bf16* b) {
    v8bf alo = *(const v8bf*)a;
    v8bf ahi = *(const v8bf*)(a + 16);
    v16bf av = __builtin_shufflevector(alo, ahi,
               0,1,2,3,4,5,6,7,8,9,10,11,12,13,14,15);
    v8bf blo = *(const v8bf*)b;
    v8bf bhi = *(const v8bf*)(b + 16);
    v16bf bv = __builtin_shufflevector(blo, bhi,
               0,1,2,3,4,5,6,7,8,9,10,11,12,13,14,15);
    return __builtin_amdgcn_wmma_f32_16x16x32_bf16(
               false, av, false, bv, (short)0, acc, false, false);
}

// Two independent accumulator chains over K (even/odd 32-chunks) to keep two
// WMMAs in flight per wave.
__device__ __forceinline__ void dual_loop(v8f& acc0, v8f& acc1,
                                          const __bf16* aRow, const __bf16* bRow,
                                          int K) {
#pragma unroll 2
    for (int k0 = 0; k0 < K; k0 += 64) {
        acc0 = wmma_step(acc0, aRow + k0,      bRow + k0);
        acc1 = wmma_step(acc1, aRow + k0 + 32, bRow + k0 + 32);
    }
}

// ---------------------------------------------------------------------------
// Group GEMM: out(16 x N) = act(concat[A0(16xK0), A1(16xK1)] @ W + bias).
// The group's 64 workers (8 blocks x 8 waves) round-robin the N/16 tiles.
// A operands read directly from global (L2-resident group buffers).
// ---------------------------------------------------------------------------
__device__ __forceinline__ void gemm_grp(const __bf16* A0, int K0,
                                         const __bf16* A1, int K1,
                                         const __bf16* Wt, const float* bias, int N,
                                         __bf16* outB, float* outF,
                                         bool relu, int worker, int lane) {
    const int m  = lane & 15;
    const int kh = lane >> 4;
    const int K  = K0 + K1;
    const __bf16* a0 = A0 + (size_t)m * K0 + kh * 8;
    const __bf16* a1 = A1 ? (A1 + (size_t)m * K1 + kh * 8) : nullptr;
    const int ntiles = N >> 4;
    for (int ti = worker; ti < ntiles; ti += NWORKERS_) {
        const int n0 = ti << 4;
        const __bf16* bRow = Wt + (size_t)(n0 + m) * K + kh * 8;
        v8f acc0 = {}, acc1 = {};
        dual_loop(acc0, acc1, a0, bRow, K0);
        if (a1) dual_loop(acc0, acc1, a1, bRow + K0, K1);
        v8f acc = acc0 + acc1;
        // C/D layout: VGPR r -> row kh*8+r, col = n0 + (lane&15).
        const int col = n0 + m;
        const int rb  = kh * 8;
        const float bv = bias[col];
#pragma unroll
        for (int r = 0; r < 8; ++r) {
            float v = acc[r] + bv;
            if (relu) v = fmaxf(v, 0.0f);
            if (outB) outB[(size_t)(rb + r) * N + col] = (__bf16)v;
            if (outF) outF[(size_t)(rb + r) * N + col] = v;
        }
    }
}

__device__ __forceinline__ float sigf(float x) {
    return 1.0f / (1.0f + __expf(-x));
}

// Monotonic 8-block group barrier (counter pre-zeroed each call).
__device__ __forceinline__ void group_barrier(unsigned* cnt, unsigned& target) {
    __syncthreads();
    target += NSLICE_;
    if (threadIdx.x == 0) {
        __hip_atomic_fetch_add(cnt, 1u, __ATOMIC_RELEASE, __HIP_MEMORY_SCOPE_AGENT);
        while (__hip_atomic_load(cnt, __ATOMIC_ACQUIRE, __HIP_MEMORY_SCOPE_AGENT) < target)
            __builtin_amdgcn_s_sleep(2);
    }
    __syncthreads();
}

struct VrnnParams {
    const float *x, *eps_z, *eps_x;
    const float *ex_b1, *ex_b2, *enc_b1, *enc_b2, *ez_b1, *ez_b2, *dec_b1, *dec_b2;
    const float *b_ih, *b_hh;
    const __bf16 *ex_w1t, *ex_w2t, *enc_w1t, *enc_w2t, *ez_w1t, *ez_w2t;
    const __bf16 *dec_w1t, *dec_w2t, *w_iht, *w_hht;
    __bf16* actB; float* actF; unsigned* bars;
    float *outRec, *outZ, *outX;
};

// ---------------------------------------------------------------------------
// Persistent VRNN scan: 128 blocks = 16 groups x 8 N-slices. Zero LDS; all
// cross-block activation sharing goes through tiny L2-resident buffers.
// ---------------------------------------------------------------------------
__global__ __launch_bounds__(256, 1) void vrnn_kernel(VrnnParams p) {
    const int tid    = threadIdx.x;
    const int wave   = tid >> 5;
    const int lane   = tid & 31;
    const int grp    = blockIdx.x >> 3;         // batch-tile group
    const int j      = blockIdx.x & (NSLICE_ - 1);
    const int worker = j * 8 + wave;            // 0..63 tile worker in group
    const int gtid   = j * 256 + tid;           // 0..2047 group thread
    const int b0     = grp * 16;

    __bf16* gb = p.actB + (size_t)grp * GA_BF16_TOTAL;
    float*  gf = p.actF + (size_t)grp * GF_TOTAL;
    unsigned* cnt = p.bars + grp * 32;

    __bf16 *xin = gb + GA_XIN,  *hid = gb + GA_HID,   *embx = gb + GA_EMBX;
    __bf16 *embz = gb + GA_EMBZ, *embxh = gb + GA_EMBXH;
    __bf16 *zb = gb + GA_Z,     *xhb = gb + GA_XH;
    __bf16 *h0 = gb + GA_H0,    *h1 = gb + GA_H1;
    float  *cst = gf + GF_C,    *fst = gf + GF_F;

    unsigned target = 0;
    const int m  = lane & 15;
    const int kh = lane >> 4;

#pragma unroll 1
    for (int t = 0; t < T_; ++t) {
        __bf16* hcur = (t & 1) ? h1 : h0;
        __bf16* hnxt = (t & 1) ? h0 : h1;

        // P1: stage x_t -> bf16
        for (int i = gtid; i < 16 * I_; i += GTHREADS_) {
            int r = i / I_, c = i - r * I_;
            xin[i] = (__bf16)p.x[((size_t)(b0 + r) * T_ + t) * I_ + c];
        }
        group_barrier(cnt, target);

        // emb_x = MLP_ex(x_t)
        gemm_grp(xin, I_, nullptr, 0, p.ex_w1t, p.ex_b1, H_, hid, nullptr, true,  worker, lane);
        group_barrier(cnt, target);
        gemm_grp(hid, H_, nullptr, 0, p.ex_w2t, p.ex_b2, H_, embx, nullptr, false, worker, lane);
        group_barrier(cnt, target);

        // enc MLP on concat[emb_x, h]
        gemm_grp(embx, H_, hcur, H_, p.enc_w1t, p.enc_b1, H_, hid, nullptr, true, worker, lane);
        group_barrier(cnt, target);
        gemm_grp(hid, H_, nullptr, 0, p.enc_w2t, p.enc_b2, 2 * L_, nullptr, fst, false, worker, lane);
        group_barrier(cnt, target);

        // reparameterize z; emit z_params
        for (int i = gtid; i < 16 * L_; i += GTHREADS_) {
            int r = i / L_, l = i - r * L_;
            float zmu = fst[r * 2 * L_ + l];
            float zlv = fst[r * 2 * L_ + L_ + l];
            size_t ob = ((size_t)(b0 + r) * T_ + t) * (2 * L_);
            p.outZ[ob + l]      = zmu;
            p.outZ[ob + L_ + l] = zlv;
            float z = zmu + __expf(0.5f * zlv) * p.eps_z[((size_t)(b0 + r) * T_ + t) * L_ + l];
            zb[r * L_ + l] = (__bf16)z;
        }
        group_barrier(cnt, target);

        // emb_z = MLP_ez(z)
        gemm_grp(zb, L_, nullptr, 0, p.ez_w1t, p.ez_b1, H_, hid, nullptr, true,  worker, lane);
        group_barrier(cnt, target);
        gemm_grp(hid, H_, nullptr, 0, p.ez_w2t, p.ez_b2, H_, embz, nullptr, false, worker, lane);
        group_barrier(cnt, target);

        // dec MLP on concat[emb_z, h]
        gemm_grp(embz, H_, hcur, H_, p.dec_w1t, p.dec_b1, H_, hid, nullptr, true, worker, lane);
        group_barrier(cnt, target);
        gemm_grp(hid, H_, nullptr, 0, p.dec_w2t, p.dec_b2, 2 * I_, nullptr, fst, false, worker, lane);
        group_barrier(cnt, target);

        // x_hat; emit x_params + reconstr
        for (int i = gtid; i < 16 * I_; i += GTHREADS_) {
            int r = i / I_, c = i - r * I_;
            float xmu = fst[r * 2 * I_ + c];
            float xlv = fst[r * 2 * I_ + I_ + c];
            size_t ob = ((size_t)(b0 + r) * T_ + t) * (2 * I_);
            p.outX[ob + c]      = xmu;
            p.outX[ob + I_ + c] = xlv;
            float xs = fminf(__expf(0.5f * xlv), 100.0f);
            float v  = xmu + xs * p.eps_x[((size_t)(b0 + r) * T_ + t) * I_ + c];
            p.outRec[((size_t)(b0 + r) * T_ + t) * I_ + c] = v;
            xhb[r * I_ + c] = (__bf16)v;
        }
        group_barrier(cnt, target);

        // emb_xhat = MLP_ex(x_hat)
        gemm_grp(xhb, I_, nullptr, 0, p.ex_w1t, p.ex_b1, H_, hid, nullptr, true,  worker, lane);
        group_barrier(cnt, target);
        gemm_grp(hid, H_, nullptr, 0, p.ex_w2t, p.ex_b2, H_, embxh, nullptr, false, worker, lane);
        group_barrier(cnt, target);

        // LSTM: worker owns 16 hidden units; 4 independent gate WMMA chains.
        {
            const int u0 = worker * 16;
            const __bf16* aX = embxh + (size_t)m * H_ + kh * 8;
            const __bf16* aZ = embz  + (size_t)m * H_ + kh * 8;
            const __bf16* aH = hcur  + (size_t)m * H_ + kh * 8;
            float g[4][8];
#pragma unroll
            for (int gi = 0; gi < 4; ++gi) {
                v8f a0 = {}, a1 = {};
                const __bf16* bI = p.w_iht + (size_t)(gi * H_ + u0 + m) * (2 * H_) + kh * 8;
                dual_loop(a0, a1, aX, bI, H_);          // K 0..1023  (emb_xhat)
                dual_loop(a0, a1, aZ, bI + H_, H_);     // K 1024..2047 (emb_z)
                const __bf16* bH = p.w_hht + (size_t)(gi * H_ + u0 + m) * H_ + kh * 8;
                dual_loop(a0, a1, aH, bH, H_);          // + h @ w_hh
                v8f acc = a0 + a1;
#pragma unroll
                for (int r = 0; r < 8; ++r) g[gi][r] = acc[r];
            }
            const int u = u0 + m, rb = kh * 8;
#pragma unroll
            for (int r = 0; r < 8; ++r) {
                const int row = rb + r;
                float ig = g[0][r] + p.b_ih[u]           + p.b_hh[u];
                float fg = g[1][r] + p.b_ih[H_ + u]      + p.b_hh[H_ + u];
                float gv = g[2][r] + p.b_ih[2 * H_ + u]  + p.b_hh[2 * H_ + u];
                float og = g[3][r] + p.b_ih[3 * H_ + u]  + p.b_hh[3 * H_ + u];
                float co = cst[row * H_ + u];
                float cn = sigf(fg) * co + sigf(ig) * tanhf(gv);
                float hn = sigf(og) * tanhf(cn);
                cst[row * H_ + u]  = cn;
                hnxt[row * H_ + u] = (__bf16)hn;
            }
        }
        group_barrier(cnt, target);
    }
}

// ---------------------------------------------------------------------------
// Host launch
// ---------------------------------------------------------------------------
extern "C" void kernel_launch(void* const* d_in, const int* in_sizes, int n_in,
                              void* d_out, int out_size, void* d_ws, size_t ws_size,
                              hipStream_t stream) {
    (void)in_sizes; (void)n_in; (void)out_size; (void)ws_size;

    __bf16* ws = (__bf16*)d_ws;
    size_t off = 0;
    auto conv = [&](int idx, int K, int N) -> const __bf16* {
        __bf16* dst = ws + off;
        int total = K * N;
        wcvt_kernel<<<(total + 255) / 256, 256, 0, stream>>>(
            (const float*)d_in[idx], dst, K, N);
        off += (size_t)total;
        return dst;
    };

    VrnnParams p;
    p.x      = (const float*)d_in[0];
    p.eps_z  = (const float*)d_in[1];
    p.eps_x  = (const float*)d_in[2];
    p.ex_b1  = (const float*)d_in[4];
    p.ex_b2  = (const float*)d_in[6];
    p.enc_b1 = (const float*)d_in[8];
    p.enc_b2 = (const float*)d_in[10];
    p.ez_b1  = (const float*)d_in[12];
    p.ez_b2  = (const float*)d_in[14];
    p.dec_b1 = (const float*)d_in[16];
    p.dec_b2 = (const float*)d_in[18];
    p.b_ih   = (const float*)d_in[20];
    p.b_hh   = (const float*)d_in[22];

    p.ex_w1t  = conv(3,  I_,     H_);       // 128 x 1024
    p.ex_w2t  = conv(5,  H_,     H_);       // 1024 x 1024
    p.enc_w1t = conv(7,  2 * H_, H_);       // 2048 x 1024
    p.enc_w2t = conv(9,  H_,     2 * L_);   // 1024 x 512
    p.ez_w1t  = conv(11, L_,     H_);       // 256 x 1024
    p.ez_w2t  = conv(13, H_,     H_);       // 1024 x 1024
    p.dec_w1t = conv(15, 2 * H_, H_);       // 2048 x 1024
    p.dec_w2t = conv(17, H_,     2 * I_);   // 1024 x 256
    p.w_iht   = conv(19, 2 * H_, 4 * H_);   // 2048 x 4096
    p.w_hht   = conv(21, H_,     4 * H_);   // 1024 x 4096
    // off == WOFF_TOTAL (20,054,016 bf16 elements)

    p.actB = ws + WOFF_TOTAL;
    p.actF = (float*)(ws + WOFF_TOTAL + (size_t)NGROUPS_ * GA_BF16_TOTAL);
    p.bars = (unsigned*)(p.actF + (size_t)NGROUPS_ * GF_TOTAL);

    init_ws_kernel<<<(NGROUPS_ * 16 * H_ + 255) / 256, 256, 0, stream>>>(
        p.actB, p.actF, p.bars);

    float* out = (float*)d_out;
    p.outRec = out;
    p.outZ   = out + (size_t)B_ * T_ * I_;
    p.outX   = out + (size_t)B_ * T_ * I_ + (size_t)B_ * T_ * 2 * L_;

    vrnn_kernel<<<dim3(NGROUPS_ * NSLICE_), dim3(256), 0, stream>>>(p);
}
